// MultiHeadsAttention_23141283791548
// MI455X (gfx1250) — matrix operand order repaired and checked by reading, not compiled, not run on hardware
//
#include <hip/hip_runtime.h>

// ---------------------------------------------------------------------------
// MI455X (gfx1250) fused causal MHA:
//   1) Q = rope(subject@wq + bq), K = rope(object@wk + bk), V = object@wv + bv
//      via bf16 WMMA GEMM with fused bias + RoPE epilogue (f32 accum)
//   2) flash-attention per (b, h, 128-query tile), bf16 WMMA for QK^T and PV
//   3) out = attn@wc + bc via the same WMMA GEMM (f32 output)
// Workspace: 4 x 32MB bf16 tensors (q, k, v, attn_out), [B*S, D] layout.
// LDS layouts chosen so every WMMA fragment read is 2x contiguous 16B
// (ds_load_b128); B/V are stored transposed in LDS to achieve this.
// Async global->LDS copies (gfx1250 GLOBAL_LOAD_ASYNC_TO_LDS_B128) used when
// the toolchain exposes the builtin; falls back to uint4 copies otherwise.
// ---------------------------------------------------------------------------

typedef __bf16          v16bf __attribute__((ext_vector_type(16)));
typedef float           v8f   __attribute__((ext_vector_type(8)));
typedef unsigned short  v16us __attribute__((ext_vector_type(16)));
typedef unsigned short  v8us  __attribute__((ext_vector_type(8)));
typedef int             v4i   __attribute__((ext_vector_type(4)));

__device__ __forceinline__ unsigned short f2bf(float f) {
  unsigned int u = __builtin_bit_cast(unsigned int, f);
  u += 0x7FFFu + ((u >> 16) & 1u);              // round-to-nearest-even
  return (unsigned short)(u >> 16);
}
__device__ __forceinline__ v16bf us2bf(v16us v) {
  return __builtin_bit_cast(v16bf, v);
}
// Two contiguous 16B runs -> one 16-element fragment.
__device__ __forceinline__ v16us ld_frag2(const unsigned short* p0,
                                          const unsigned short* p1) {
  v8us a = *(const v8us*)p0;
  v8us b = *(const v8us*)p1;
  return __builtin_shufflevector(a, b, 0, 1, 2, 3, 4, 5, 6, 7,
                                 8, 9, 10, 11, 12, 13, 14, 15);
}

// ---- gfx1250 async global->LDS copy (guarded probe) -----------------------
// Diagnostic from the previous round shows the builtin's parameters are
// pointers to 128-bit int vectors in the generic address space.
#if defined(__has_builtin)
#if __has_builtin(__builtin_amdgcn_global_load_async_to_lds_b128)
#define HAS_ASYNC_COPY 1
#endif
#endif

#ifdef HAS_ASYNC_COPY
__device__ __forceinline__ void async_copy16(const void* g, void* l) {
  __builtin_amdgcn_global_load_async_to_lds_b128((v4i*)g, (v4i*)l, 0, 0);
}
__device__ __forceinline__ void wait_async0() {
#if __has_builtin(__builtin_amdgcn_s_wait_asynccnt)
  __builtin_amdgcn_s_wait_asynccnt(0);
#else
  asm volatile("s_wait_asynccnt 0" ::: "memory");
#endif
}
#endif

// ---------------------------------------------------------------------------
// Generic tiled GEMM:  Out[M,N] = A[M,K] @ W[K,N] + bias, optional RoPE.
// Block: 256 threads (8 waves), tile 128x128, K-step 32.
// Waves: 4 row-groups (32 rows) x 2 col-groups (64 cols) -> 2x4 WMMA C tiles.
// Bst holds the W tile transposed so B-fragment reads are contiguous in k.
// ---------------------------------------------------------------------------
template<bool ROPE, bool OUT_F32, bool IN_BF16>
__global__ __launch_bounds__(256, 1)
void gemm_rope_kernel(const void* __restrict__ Ain,
                      const float* __restrict__ W,
                      const float* __restrict__ bias,
                      void* __restrict__ Out,
                      int M, int N, int K, int S,
                      const float* __restrict__ base_ptr)
{
  constexpr int BM = 128, BN = 128, BK = 32;
  constexpr int LDA = BK + 8;    // 40 us = 80B rows (16B multiple)
  __shared__ alignas(16) unsigned short As[BM][LDA];
  __shared__ alignas(16) unsigned short Bst[BN][LDA];   // transposed: [n][k]

  const int tid  = threadIdx.x;
  const int lane = tid & 31;
  const int hf   = lane >> 4;    // 0/1: lane half (wave32)
  const int l15  = lane & 15;
  const int wave = tid >> 5;
  const int wm   = wave & 3;     // row group (32 rows each)
  const int wn   = wave >> 2;    // col group (64 cols each)
  const int m0   = blockIdx.y * BM;
  const int n0   = blockIdx.x * BN;

  const float base    = ROPE ? base_ptr[0] : 1.0f;
  const float logbase = ROPE ? __logf(base) : 0.0f;

  v8f zero = {};
  v8f acc[2][4];
#pragma unroll
  for (int a = 0; a < 2; ++a)
#pragma unroll
    for (int b = 0; b < 4; ++b) acc[a][b] = zero;

  const int ar = tid >> 1, akc = (tid & 1) * 16;  // A fill: row, 16-wide k chunk
  const int br = tid >> 3, bnc = (tid & 7) * 16;  // B fill: k row, 16-wide n chunk

  for (int k0 = 0; k0 < K; k0 += BK) {
    // ---- stage A tile (BM x BK) as bf16 ----
    if (IN_BF16) {
      const unsigned short* g =
          (const unsigned short*)Ain + (size_t)(m0 + ar) * K + k0 + akc;
#ifdef HAS_ASYNC_COPY
      async_copy16(g,     &As[ar][akc]);
      async_copy16(g + 8, &As[ar][akc + 8]);
#else
      *(uint4*)&As[ar][akc]     = *(const uint4*)g;
      *(uint4*)&As[ar][akc + 8] = *(const uint4*)(g + 8);
#endif
    } else {
      const float4* src =
          (const float4*)((const float*)Ain + (size_t)(m0 + ar) * K + k0 + akc);
#pragma unroll
      for (int c = 0; c < 4; ++c) {
        float4 f = src[c];
        unsigned int lo = (unsigned)f2bf(f.x) | ((unsigned)f2bf(f.y) << 16);
        unsigned int hi = (unsigned)f2bf(f.z) | ((unsigned)f2bf(f.w) << 16);
        *(unsigned int*)&As[ar][akc + c * 4]     = lo;
        *(unsigned int*)&As[ar][akc + c * 4 + 2] = hi;
      }
    }
    // ---- stage W tile (BK x BN) transposed into Bst[n][k] as bf16 ----
    {
      const float4* src = (const float4*)(W + (size_t)(k0 + br) * N + n0 + bnc);
#pragma unroll
      for (int c = 0; c < 4; ++c) {
        float4 f = src[c];
        Bst[bnc + c * 4 + 0][br] = f2bf(f.x);
        Bst[bnc + c * 4 + 1][br] = f2bf(f.y);
        Bst[bnc + c * 4 + 2][br] = f2bf(f.z);
        Bst[bnc + c * 4 + 3][br] = f2bf(f.w);
      }
    }
    // prefetch next K-step tiles (gfx1250 global_prefetch_b8)
    if (k0 + BK < K) {
      if (IN_BF16)
        __builtin_prefetch((const unsigned short*)Ain +
                           (size_t)(m0 + ar) * K + k0 + BK + akc, 0, 1);
      else
        __builtin_prefetch((const float*)Ain +
                           (size_t)(m0 + ar) * K + k0 + BK + akc, 0, 1);
      __builtin_prefetch(W + (size_t)(k0 + BK + br) * N + n0 + bnc, 0, 1);
    }
#ifdef HAS_ASYNC_COPY
    if constexpr (IN_BF16) wait_async0();
#endif
    __syncthreads();

    // ---- fragments: all reads are 2x contiguous 16B ----
    v16us av[2];
#pragma unroll
    for (int tr = 0; tr < 2; ++tr) {
      const int row = wm * 32 + tr * 16 + l15;          // A: M = lane%16
      av[tr] = ld_frag2(&As[row][hf * 8], &As[row][16 + hf * 8]);
    }
    v16us bvf[4];
#pragma unroll
    for (int tc = 0; tc < 4; ++tc) {
      const int col = wn * 64 + tc * 16 + l15;          // B: N = lane%16
      bvf[tc] = ld_frag2(&Bst[col][hf * 8], &Bst[col][16 + hf * 8]);
    }
#pragma unroll
    for (int tr = 0; tr < 2; ++tr)
#pragma unroll
      for (int tc = 0; tc < 4; ++tc)
        acc[tr][tc] = __builtin_amdgcn_wmma_f32_16x16x32_bf16(
            false, us2bf(av[tr]), false, us2bf(bvf[tc]),
            (short)0, acc[tr][tc], false, false);
    __syncthreads();
  }

  // ---- epilogue: bias (+RoPE) + store ----
#pragma unroll
  for (int tr = 0; tr < 2; ++tr) {
#pragma unroll
    for (int tc = 0; tc < 4; ++tc) {
      const int colg = n0 + wn * 64 + tc * 16 + l15;
      const float bval = bias[colg];
#pragma unroll
      for (int i = 0; i < 8; ++i) {
        const int rowg = m0 + wm * 32 + tr * 16 + hf * 8 + i;  // C: M = 8*half + i
        float x = acc[tr][tc][i] + bval;
        float outv = x;
        if (ROPE) {
          // pair partner (col^1) lives in lane^1 (same half)
          float p = __shfl_xor(x, 1);
          const int fi = (colg & 127) >> 1;                 // freq idx within head
          const float inv = __expf(-(float)fi * (1.0f / 64.0f) * logbase);
          const float ang = (float)(rowg % S) * inv;
          float sv, cv;
          __sincosf(ang, &sv, &cv);
          outv = (colg & 1) ? (p * sv + x * cv) : (x * cv - p * sv);
        }
        if (OUT_F32)
          ((float*)Out)[(size_t)rowg * N + colg] = outv;
        else
          ((unsigned short*)Out)[(size_t)rowg * N + colg] = f2bf(outv);
      }
    }
  }
}

// ---------------------------------------------------------------------------
// Flash attention. Block = 256 threads (8 waves), one (b, h, 128-query tile).
// Each wave owns 16 query rows: Q frags in registers, online softmax,
// O accum 16x128 in f32 registers. K tile row-major (async-fillable);
// V tile stored transposed (Vt[hd][t]) so PV B-fragments read contiguously.
// ---------------------------------------------------------------------------
__global__ __launch_bounds__(256, 1)
void attn_kernel(const unsigned short* __restrict__ Q,
                 const unsigned short* __restrict__ Kg,
                 const unsigned short* __restrict__ Vg,
                 unsigned short* __restrict__ O,
                 int S, int D)
{
  __shared__ alignas(16) unsigned short Ks[64][128 + 8];   // [t][hd]
  __shared__ alignas(16) unsigned short Vt[128][64 + 8];   // [hd][t] transposed
  __shared__ alignas(16) unsigned short Ps[8][16][64 + 8]; // per-wave P staging

  const int tid  = threadIdx.x;
  const int lane = tid & 31;
  const int hf   = lane >> 4;
  const int l15  = lane & 15;
  const int wave = tid >> 5;

  const int qt = blockIdx.x, h = blockIdx.y, b = blockIdx.z;
  const int q0 = qt * 128;
  const size_t rowbase = (size_t)b * S;
  const int hbase = h * 128;
  const int qrow0 = q0 + wave * 16;

  // Q fragments (A-layout), 4 K-steps across HD=128, kept in registers
  v16us qf[4];
  {
    const unsigned short* qp = Q + (rowbase + qrow0 + l15) * D + hbase;
#pragma unroll
    for (int kk = 0; kk < 4; ++kk)
      qf[kk] = ld_frag2(qp + kk * 32 + hf * 8, qp + kk * 32 + 16 + hf * 8);
  }

  v8f zero = {};
  v8f o_acc[8];
#pragma unroll
  for (int j = 0; j < 8; ++j) o_acc[j] = zero;
  float m_i[8], l_i[8];
#pragma unroll
  for (int i = 0; i < 8; ++i) { m_i[i] = -3.0e38f; l_i[i] = 0.0f; }

  const float scale = 0.08838834764831845f;  // 1/sqrt(128)
  const int nkv = 2 * qt + 2;                // causal: kv tiles up to diagonal

  for (int t = 0; t < nkv; ++t) {
    const int t0 = t * 64;
    // ---- cooperative K/V tile load (64 x 128 bf16 each) ----
#pragma unroll
    for (int c = 0; c < 4; ++c) {
      const int idx = tid + c * 256;
      const int r  = idx >> 4;            // t-local 0..63
      const int cc = (idx & 15) * 8;      // hd chunk
      const size_t g = (rowbase + t0 + r) * (size_t)D + hbase + cc;
#ifdef HAS_ASYNC_COPY
      async_copy16(Kg + g, &Ks[r][cc]);   // K: async DMA, row-major
#else
      *(uint4*)&Ks[r][cc] = *(const uint4*)(Kg + g);
#endif
      v8us vv = *(const v8us*)(Vg + g);   // V: transpose into Vt[hd][t]
#pragma unroll
      for (int e = 0; e < 8; ++e) Vt[cc + e][r] = vv[e];
    }
#ifdef HAS_ASYNC_COPY
    wait_async0();
#endif
    __syncthreads();

    // ---- S = Q K^T  (16 x 64 per wave) ----
    v8f sf[4];
#pragma unroll
    for (int j = 0; j < 4; ++j) sf[j] = zero;
#pragma unroll
    for (int kk = 0; kk < 4; ++kk) {
#pragma unroll
      for (int j = 0; j < 4; ++j) {
        // B[k][n] = K[n][k]: contiguous along hd in Ks rows
        v16us bf = ld_frag2(&Ks[j * 16 + l15][kk * 32 + hf * 8],
                            &Ks[j * 16 + l15][kk * 32 + 16 + hf * 8]);
        sf[j] = __builtin_amdgcn_wmma_f32_16x16x32_bf16(
            false, us2bf(qf[kk]), false, us2bf(bf), (short)0, sf[j], false, false);
      }
    }

    // ---- scale + causal mask + online softmax (rows in 16-lane halves) ----
#pragma unroll
    for (int i = 0; i < 8; ++i) {
      const int srow = qrow0 + hf * 8 + i;
      float rmax = -3.0e38f;
#pragma unroll
      for (int j = 0; j < 4; ++j) {
        const int tcol = t0 + j * 16 + l15;
        float x = sf[j][i] * scale;
        x = (tcol <= srow) ? x : -1.0e30f;
        sf[j][i] = x;
        rmax = fmaxf(rmax, x);
      }
#pragma unroll
      for (int off = 1; off < 16; off <<= 1)
        rmax = fmaxf(rmax, __shfl_xor(rmax, off));
      const float mnew  = fmaxf(m_i[i], rmax);
      const float alpha = __expf(m_i[i] - mnew);
      float rsum = 0.0f;
#pragma unroll
      for (int j = 0; j < 4; ++j) {
        const float p = __expf(sf[j][i] - mnew);
        sf[j][i] = p;
        rsum += p;
      }
#pragma unroll
      for (int off = 1; off < 16; off <<= 1) rsum += __shfl_xor(rsum, off);
      l_i[i] = l_i[i] * alpha + rsum;
      m_i[i] = mnew;
#pragma unroll
      for (int j = 0; j < 8; ++j) o_acc[j][i] *= alpha;
    }

    // ---- P through LDS (C-layout -> A-layout), then O += P V ----
#pragma unroll
    for (int j = 0; j < 4; ++j)
#pragma unroll
      for (int i = 0; i < 8; ++i)
        Ps[wave][hf * 8 + i][j * 16 + l15] = f2bf(sf[j][i]);
    // same-wave LDS ops stay in order; compiler inserts s_wait_dscnt

#pragma unroll
    for (int kk = 0; kk < 2; ++kk) {
      v16us pa = ld_frag2(&Ps[wave][l15][kk * 32 + hf * 8],
                          &Ps[wave][l15][kk * 32 + 16 + hf * 8]);
#pragma unroll
      for (int j = 0; j < 8; ++j) {
        // B[k=t][n=hd] = Vt[hd][t]: contiguous along t in Vt rows
        v16us vb = ld_frag2(&Vt[j * 16 + l15][kk * 32 + hf * 8],
                            &Vt[j * 16 + l15][kk * 32 + 16 + hf * 8]);
        o_acc[j] = __builtin_amdgcn_wmma_f32_16x16x32_bf16(
            false, us2bf(pa), false, us2bf(vb), (short)0, o_acc[j], false, false);
      }
    }
    __syncthreads();
  }

  // ---- normalize + store attention output (bf16, [B*S, D]) ----
#pragma unroll
  for (int j = 0; j < 8; ++j)
#pragma unroll
    for (int i = 0; i < 8; ++i) {
      const float v = o_acc[j][i] / l_i[i];
      O[(rowbase + qrow0 + hf * 8 + i) * (size_t)D + hbase + j * 16 + l15] = f2bf(v);
    }
}

// ---------------------------------------------------------------------------
extern "C" void kernel_launch(void* const* d_in, const int* in_sizes, int n_in,
                              void* d_out, int out_size, void* d_ws, size_t ws_size,
                              hipStream_t stream)
{
  (void)in_sizes; (void)n_in; (void)out_size; (void)ws_size;
  const int B = 4, S = 2048, D = 2048, H = 16;
  const int M = B * S;  // 8192

  const float* subj = (const float*)d_in[0];
  const float* obj  = (const float*)d_in[1];
  const float* wq   = (const float*)d_in[2];
  const float* bq   = (const float*)d_in[3];
  const float* wk   = (const float*)d_in[4];
  const float* bk   = (const float*)d_in[5];
  const float* wv   = (const float*)d_in[6];
  const float* bv   = (const float*)d_in[7];
  const float* wc   = (const float*)d_in[8];
  const float* bc   = (const float*)d_in[9];
  const float* rb   = (const float*)d_in[10];
  // d_in[11] (msk) unused: causal mask applied analytically.

  const size_t tbytes = (size_t)M * D * sizeof(unsigned short);  // 32 MB each
  unsigned short* q_ws = (unsigned short*)d_ws;
  unsigned short* k_ws = (unsigned short*)((char*)d_ws + tbytes);
  unsigned short* v_ws = (unsigned short*)((char*)d_ws + 2 * tbytes);
  unsigned short* o_ws = (unsigned short*)((char*)d_ws + 3 * tbytes);

  dim3 blk(256);
  dim3 ggrid(D / 128, M / 128);  // 16 x 64

  gemm_rope_kernel<true,  false, false><<<ggrid, blk, 0, stream>>>(
      subj, wq, bq, q_ws, M, D, D, S, rb);
  gemm_rope_kernel<true,  false, false><<<ggrid, blk, 0, stream>>>(
      obj,  wk, bk, k_ws, M, D, D, S, rb);
  gemm_rope_kernel<false, false, false><<<ggrid, blk, 0, stream>>>(
      obj,  wv, bv, v_ws, M, D, D, S, rb);

  dim3 agrid(S / 128, H, B);     // 16 x 16 x 4
  attn_kernel<<<agrid, blk, 0, stream>>>(q_ws, k_ws, v_ws, o_ws, S, D);

  gemm_rope_kernel<false, true, true><<<ggrid, blk, 0, stream>>>(
      o_ws, wc, bc, (float*)d_out, M, D, D, S, rb);
}
